// GATModel_83580063580841
// MI455X (gfx1250) — compile-verified, matrix operand order checked by reading, not run on hardware
//
#include <hip/hip_runtime.h>
#include <cstdint>
#include <cstddef>

// ---------------------------------------------------------------------------
// GAT 3-layer model for MI455X (gfx1250), wave32, WMMA bf16 GEMMs,
// TDM (tensor_load_to_lds) weight staging.
// ---------------------------------------------------------------------------

#define NEG_SLOPE 0.2f

typedef __bf16 bf16x16 __attribute__((ext_vector_type(16)));
typedef float  f32x8   __attribute__((ext_vector_type(8)));
typedef unsigned u32x4 __attribute__((ext_vector_type(4)));
typedef int      i32x4 __attribute__((ext_vector_type(4)));
typedef int      i32x8 __attribute__((ext_vector_type(8)));

union Frag16 { bf16x16 v; uint4 q[2]; };

__device__ __forceinline__ unsigned short f2bf(float f) {
    unsigned u = __float_as_uint(f);
    u += 0x7FFFu + ((u >> 16) & 1u);   // round-to-nearest-even
    return (unsigned short)(u >> 16);
}

// order-preserving float <-> uint mapping for atomic max (memset-0 == -NaN)
__device__ __forceinline__ unsigned fmap(float f) {
    unsigned u = __float_as_uint(f);
    return (u & 0x80000000u) ? ~u : (u | 0x80000000u);
}
__device__ __forceinline__ float funmap(unsigned u) {
    return (u & 0x80000000u) ? __uint_as_float(u ^ 0x80000000u)
                             : __uint_as_float(~u);
}

__device__ __forceinline__ void atomAddF(float* p, float v) {
    __hip_atomic_fetch_add(p, v, __ATOMIC_RELAXED, __HIP_MEMORY_SCOPE_AGENT);
}

__device__ __forceinline__ float lrelu(float x) {
    return (x > 0.0f) ? x : NEG_SLOPE * x;
}

// ---------------------------------------------------------------------------
// utility: zero fill (uint4 granularity), graph-capture safe
// ---------------------------------------------------------------------------
__global__ void zero4_kernel(uint4* __restrict__ p, long long n4) {
    long long i = (long long)blockIdx.x * blockDim.x + threadIdx.x;
    long long stride = (long long)gridDim.x * blockDim.x;
    uint4 z = make_uint4(0u, 0u, 0u, 0u);
    for (; i < n4; i += stride) p[i] = z;
}

// ---------------------------------------------------------------------------
// convert W [128(k) x 128(n)] fp32 row-major -> Wt [n][k] bf16
// ---------------------------------------------------------------------------
__global__ void wt_convert_kernel(const float* __restrict__ W,
                                  unsigned short* __restrict__ Wt) {
    int t = blockIdx.x * blockDim.x + threadIdx.x;   // 16384 threads
    int n = t >> 7, k = t & 127;
    Wt[n * 128 + k] = f2bf(W[k * 128 + n]);
}

// ---------------------------------------------------------------------------
// GEMM: C[N x 128] = A[N x 128] * W[128 x 128], via v_wmma_f32_16x16x32_bf16.
// One block: 32 rows x 128 cols; 8 waves, each wave owns one 16-col strip and
// both 16-row halves (2 accumulators, 8 WMMAs per wave).
// Wt (bf16, 32KB) is staged into LDS by the Tensor Data Mover, overlapping
// with the fp32->bf16 conversion of the A tile on the VALU.
// ---------------------------------------------------------------------------
__global__ __launch_bounds__(256)
void gemm_wmma_kernel(const float* __restrict__ A,
                      const unsigned short* __restrict__ Wt,   // bf16 [n][k]
                      float* __restrict__ C, int Nrows) {
    __shared__ unsigned short sA[32 * 128];    // 8 KB  (bf16, [m][k])
    __shared__ unsigned short sB[128 * 128];   // 32 KB (bf16, [n][k])

    const int tid  = threadIdx.x;
    const int row0 = blockIdx.x * 32;

#if __has_builtin(__builtin_amdgcn_tensor_load_to_lds)
    // ---- TDM staging of Wt: one tensor_load_to_lds issued by wave 0 ----
    if (tid < 32) {
        unsigned lds_addr = (unsigned)(uintptr_t)(void*)sB;   // LDS byte offset
        unsigned long long ga = (unsigned long long)(uintptr_t)Wt;
        // D# group 0: count=1 | lds_addr | global_addr[56:0] | type=2
        u32x4 g0;
        g0.x = 1u;                                   // count=1 (valid), user mode
        g0.y = lds_addr;                             // LDS address (bytes)
        g0.z = (unsigned)(ga & 0xFFFFFFFFu);         // global addr [31:0]
        g0.w = (unsigned)((ga >> 32) & 0x01FFFFFFu)  // global addr [56:32]
               | (2u << 30);                         // type = 2 ("image")
        // D# group 1: data_size=1 (2B); tensor 128x128; tile 128x128; stride0=128
        i32x8 g1;
        g1.s0 = (1 << 16);        // workgroup_mask=0, data_size=2B
        g1.s1 = (128 << 16);      // atomic_barrier_addr=0 | tensor_dim0[15:0]=128
        g1.s2 = (128 << 16);      // tensor_dim0[31:16]=0 | tensor_dim1[15:0]=128
        g1.s3 = (128 << 16);      // tensor_dim1[31:16]=0 | tile_dim0=128
        g1.s4 = 128;              // tile_dim1=128, tile_dim2=0
        g1.s5 = 128;              // tensor_dim0_stride[31:0]=128
        g1.s6 = 0;                // stride0 hi | tensor_dim1_stride lo
        g1.s7 = 0;
        i32x4 g2 = {0, 0, 0, 0};  // unused (2D tile)
        i32x4 g3 = {0, 0, 0, 0};
        i32x8 g4 = {0, 0, 0, 0, 0, 0, 0, 0};   // extra group (6-arg toolchain)
        __builtin_amdgcn_tensor_load_to_lds(g0, g1, g2, g3, g4, 0);
    }
#else
    {
        const uint4* s = (const uint4*)Wt;
        uint4* d = (uint4*)sB;
        for (int i = tid; i < 2048; i += 256) d[i] = s[i];
    }
#endif

    // stage + convert A tile (32 x 128 fp32 -> bf16); clamp OOB rows to row 0
    {
        const float4* ap = (const float4*)A;
        for (int i = tid; i < 1024; i += 256) {
            int r = i >> 5;          // local row 0..31
            int c4 = i & 31;         // float4 column
            long long grow = (long long)row0 + r;
            float4 v = (grow < Nrows) ? ap[grow * 32 + c4] : ap[c4];
            int base = i * 4;
            sA[base + 0] = f2bf(v.x); sA[base + 1] = f2bf(v.y);
            sA[base + 2] = f2bf(v.z); sA[base + 3] = f2bf(v.w);
        }
    }

#if __has_builtin(__builtin_amdgcn_tensor_load_to_lds)
    if (tid < 32) __builtin_amdgcn_s_wait_tensorcnt(0);
#endif
    __syncthreads();

    const int wave = tid >> 5;
    const int lane = tid & 31;
    const int mrow = lane & 15;
    const int koff = (lane < 16) ? 0 : 8;      // lane-half K offset per ISA layout
    const int ncol = wave * 16 + mrow;

    f32x8 acc0 = {0.f,0.f,0.f,0.f,0.f,0.f,0.f,0.f};
    f32x8 acc1 = {0.f,0.f,0.f,0.f,0.f,0.f,0.f,0.f};

#pragma unroll
    for (int kk = 0; kk < 128; kk += 32) {
        Frag16 b, a0, a1;
        b.q[0]  = *(const uint4*)(sB + ncol * 128 + kk + koff);
        b.q[1]  = *(const uint4*)(sB + ncol * 128 + kk + 16 + koff);
        a0.q[0] = *(const uint4*)(sA + mrow * 128 + kk + koff);
        a0.q[1] = *(const uint4*)(sA + mrow * 128 + kk + 16 + koff);
        a1.q[0] = *(const uint4*)(sA + (16 + mrow) * 128 + kk + koff);
        a1.q[1] = *(const uint4*)(sA + (16 + mrow) * 128 + kk + 16 + koff);
        acc0 = __builtin_amdgcn_wmma_f32_16x16x32_bf16(
                   false, a0.v, false, b.v, (short)0, acc0, false, false);
        acc1 = __builtin_amdgcn_wmma_f32_16x16x32_bf16(
                   false, a1.v, false, b.v, (short)0, acc1, false, false);
    }

    // store: C/D layout — lanes 0-15: M=r, N=lane; lanes 16-31: M=8+r, N=lane-16
    const int ncs   = wave * 16 + (lane & 15);
    const int mhalf = (lane >> 4) * 8;
    if (row0 + 32 <= Nrows) {
        // fast path: 32-bit offsets, no guards (N*128 < 2^31)
        int b0 = (row0 + mhalf) * 128 + ncs;
#pragma unroll
        for (int r = 0; r < 8; ++r) {
            C[b0 + r * 128]        = acc0[r];
            C[b0 + 2048 + r * 128] = acc1[r];
        }
    } else {
#pragma unroll
        for (int r = 0; r < 8; ++r) {
            int m0 = row0 + mhalf + r;
            int m1 = m0 + 16;
            if (m0 < Nrows) C[m0 * 128 + ncs] = acc0[r];
            if (m1 < Nrows) C[m1 * 128 + ncs] = acc1[r];
        }
    }
}

// ---------------------------------------------------------------------------
// attention coefficients: a_s[n,h] = dot(xp[n,h,:], att_src[h,:]) (H=4, C=32)
// one wave per node; 8-lane group reduction per head
// ---------------------------------------------------------------------------
__global__ void attn_coef_kernel(const float* __restrict__ xp,
                                 const float* __restrict__ att_s,
                                 const float* __restrict__ att_d,
                                 float* __restrict__ as_out,
                                 float* __restrict__ ad_out, int N) {
    int node = (int)(((long long)blockIdx.x * blockDim.x + threadIdx.x) >> 5);
    int lane = threadIdx.x & 31;
    if (node >= N) return;
    float4 v  = ((const float4*)xp)[(long long)node * 32 + lane];
    float4 s4 = ((const float4*)att_s)[lane];
    float4 d4 = ((const float4*)att_d)[lane];
    float ps = v.x * s4.x + v.y * s4.y + v.z * s4.z + v.w * s4.w;
    float pd = v.x * d4.x + v.y * d4.y + v.z * d4.z + v.w * d4.w;
#pragma unroll
    for (int m = 4; m >= 1; m >>= 1) {
        ps += __shfl_xor(ps, m, 8);
        pd += __shfl_xor(pd, m, 8);
    }
    if ((lane & 7) == 0) {
        as_out[node * 4 + (lane >> 3)] = ps;
        ad_out[node * 4 + (lane >> 3)] = pd;
    }
}

// ---------------------------------------------------------------------------
// edge passes, H = 4 (layers 1 & 2). Self-loops are edges e >= E (s=d=e-E).
// ---------------------------------------------------------------------------
__global__ void edge_max_kernel(const int* __restrict__ src,
                                const int* __restrict__ dst,
                                const float* __restrict__ as_,
                                const float* __restrict__ ad_,
                                unsigned* __restrict__ m_mapped,
                                int E, int Etot) {
    int e = blockIdx.x * blockDim.x + threadIdx.x;
    if (e >= Etot) return;
    int s, d;
    if (e < E) { s = src[e]; d = dst[e]; } else { s = e - E; d = s; }
    float4 a = ((const float4*)as_)[s];
    float4 b = ((const float4*)ad_)[d];
    float al[4] = {a.x + b.x, a.y + b.y, a.z + b.z, a.w + b.w};
#pragma unroll
    for (int h = 0; h < 4; ++h)
        atomicMax(&m_mapped[d * 4 + h], fmap(lrelu(al[h])));
}

__global__ void edge_exp_kernel(const int* __restrict__ src,
                                const int* __restrict__ dst,
                                const float* __restrict__ as_,
                                const float* __restrict__ ad_,
                                const unsigned* __restrict__ m_mapped,
                                float* __restrict__ ebuf,
                                float* __restrict__ ssum,
                                int E, int Etot) {
    int e = blockIdx.x * blockDim.x + threadIdx.x;
    if (e >= Etot) return;
    int s, d;
    if (e < E) { s = src[e]; d = dst[e]; } else { s = e - E; d = s; }
    float4 a = ((const float4*)as_)[s];
    float4 b = ((const float4*)ad_)[d];
    float al[4] = {a.x + b.x, a.y + b.y, a.z + b.z, a.w + b.w};
    float4 ev;
    float* evp = (float*)&ev;
#pragma unroll
    for (int h = 0; h < 4; ++h) {
        float m = funmap(m_mapped[d * 4 + h]);
        float x = __expf(lrelu(al[h]) - m);
        evp[h] = x;
        atomAddF(&ssum[d * 4 + h], x);
    }
    ((float4*)ebuf)[e] = ev;
}

// one wave32 per edge: lane covers 4 of 128 channels, head = lane/8
__global__ __launch_bounds__(256)
void edge_aggr_kernel(const int* __restrict__ src,
                      const int* __restrict__ dst,
                      const float* __restrict__ xp,
                      const float* __restrict__ ebuf,
                      const float* __restrict__ ssum,
                      float* __restrict__ out, int E, int Etot) {
    int e = (int)(((long long)blockIdx.x * blockDim.x + threadIdx.x) >> 5);
    int lane = threadIdx.x & 31;
    if (e >= Etot) return;
    int s, d;
    if (e < E) { s = src[e]; d = dst[e]; } else { s = e - E; d = s; }
    int h = lane >> 3;
    float w = ebuf[(long long)e * 4 + h] / (ssum[d * 4 + h] + 1e-16f);
    float4 xv = ((const float4*)xp)[(long long)s * 32 + lane];
    float* o = out + (long long)d * 128 + lane * 4;
    atomAddF(o + 0, xv.x * w);
    atomAddF(o + 1, xv.y * w);
    atomAddF(o + 2, xv.z * w);
    atomAddF(o + 3, xv.w * w);
}

// ---------------------------------------------------------------------------
// epilogue: h = relu(layernorm(h + bias) * lnw + lnb), one wave per node
// ---------------------------------------------------------------------------
__global__ void ln_relu_kernel(float* __restrict__ h,
                               const float* __restrict__ bias,
                               const float* __restrict__ lnw,
                               const float* __restrict__ lnb, int N) {
    int node = (int)(((long long)blockIdx.x * blockDim.x + threadIdx.x) >> 5);
    int lane = threadIdx.x & 31;
    if (node >= N) return;
    float4 v  = ((const float4*)h)[(long long)node * 32 + lane];
    float4 b4 = ((const float4*)bias)[lane];
    v.x += b4.x; v.y += b4.y; v.z += b4.z; v.w += b4.w;
    float sum = v.x + v.y + v.z + v.w;
#pragma unroll
    for (int m = 16; m >= 1; m >>= 1) sum += __shfl_xor(sum, m, 32);
    float mean = sum * (1.0f / 128.0f);
    float dx = v.x - mean, dy = v.y - mean, dz = v.z - mean, dw = v.w - mean;
    float sq = dx * dx + dy * dy + dz * dz + dw * dw;
#pragma unroll
    for (int m = 16; m >= 1; m >>= 1) sq += __shfl_xor(sq, m, 32);
    float inv = rsqrtf(sq * (1.0f / 128.0f) + 1e-5f);
    float4 w4 = ((const float4*)lnw)[lane];
    float4 l4 = ((const float4*)lnb)[lane];
    float4 o;
    o.x = fmaxf(dx * inv * w4.x + l4.x, 0.0f);
    o.y = fmaxf(dy * inv * w4.y + l4.y, 0.0f);
    o.z = fmaxf(dz * inv * w4.z + l4.z, 0.0f);
    o.w = fmaxf(dw * inv * w4.w + l4.w, 0.0f);
    ((float4*)h)[(long long)node * 32 + lane] = o;
}

// ---------------------------------------------------------------------------
// layer 3 (H=1, OUT=1): xp3[n] = dot(h2[n,:], W3); a_s3 = xp3*as3; a_d3 = xp3*ad3
// ---------------------------------------------------------------------------
__global__ void xp3_kernel(const float* __restrict__ h,
                           const float* __restrict__ W3,
                           const float* __restrict__ as3,
                           const float* __restrict__ ad3,
                           float* __restrict__ xp3,
                           float* __restrict__ a_s,
                           float* __restrict__ a_d, int N) {
    int node = (int)(((long long)blockIdx.x * blockDim.x + threadIdx.x) >> 5);
    int lane = threadIdx.x & 31;
    if (node >= N) return;
    float4 v = ((const float4*)h)[(long long)node * 32 + lane];
    float4 w = ((const float4*)W3)[lane];
    float p = v.x * w.x + v.y * w.y + v.z * w.z + v.w * w.w;
#pragma unroll
    for (int m = 16; m >= 1; m >>= 1) p += __shfl_xor(p, m, 32);
    if (lane == 0) {
        xp3[node] = p;
        a_s[node] = p * as3[0];
        a_d[node] = p * ad3[0];
    }
}

__global__ void edge_max1_kernel(const int* __restrict__ src,
                                 const int* __restrict__ dst,
                                 const float* __restrict__ as_,
                                 const float* __restrict__ ad_,
                                 unsigned* __restrict__ m_mapped,
                                 int E, int Etot) {
    int e = blockIdx.x * blockDim.x + threadIdx.x;
    if (e >= Etot) return;
    int s, d;
    if (e < E) { s = src[e]; d = dst[e]; } else { s = e - E; d = s; }
    atomicMax(&m_mapped[d], fmap(lrelu(as_[s] + ad_[d])));
}

__global__ void edge_exp1_kernel(const int* __restrict__ src,
                                 const int* __restrict__ dst,
                                 const float* __restrict__ as_,
                                 const float* __restrict__ ad_,
                                 const unsigned* __restrict__ m_mapped,
                                 float* __restrict__ ebuf,
                                 float* __restrict__ ssum,
                                 int E, int Etot) {
    int e = blockIdx.x * blockDim.x + threadIdx.x;
    if (e >= Etot) return;
    int s, d;
    if (e < E) { s = src[e]; d = dst[e]; } else { s = e - E; d = s; }
    float x = __expf(lrelu(as_[s] + ad_[d]) - funmap(m_mapped[d]));
    ebuf[e] = x;
    atomAddF(&ssum[d], x);
}

__global__ void edge_aggr1_kernel(const int* __restrict__ src,
                                  const int* __restrict__ dst,
                                  const float* __restrict__ xp3,
                                  const float* __restrict__ ebuf,
                                  const float* __restrict__ ssum,
                                  float* __restrict__ out3,
                                  int E, int Etot) {
    int e = blockIdx.x * blockDim.x + threadIdx.x;
    if (e >= Etot) return;
    int s, d;
    if (e < E) { s = src[e]; d = dst[e]; } else { s = e - E; d = s; }
    float w = ebuf[e] / (ssum[d] + 1e-16f);
    atomAddF(&out3[d], xp3[s] * w);
}

// ---------------------------------------------------------------------------
// pooling: mean over graphs
// ---------------------------------------------------------------------------
__global__ void pool_accum_kernel(const float* __restrict__ out3,
                                  const float* __restrict__ b3,
                                  const int* __restrict__ batch,
                                  float* __restrict__ pool, int N, int G) {
    int n = blockIdx.x * blockDim.x + threadIdx.x;
    if (n >= N) return;
    float v = out3[n] + b3[0];
    int g = batch[n];
    atomAddF(&pool[g], v);
    atomAddF(&pool[G + g], 1.0f);
}

__global__ void pool_final_kernel(const float* __restrict__ pool,
                                  float* __restrict__ out, int G) {
    int g = blockIdx.x * blockDim.x + threadIdx.x;
    if (g >= G) return;
    out[g] = pool[g] / fmaxf(pool[G + g], 1.0f);
}

// ---------------------------------------------------------------------------
// host launcher
// ---------------------------------------------------------------------------
extern "C" void kernel_launch(void* const* d_in, const int* in_sizes, int n_in,
                              void* d_out, int out_size, void* d_ws, size_t ws_size,
                              hipStream_t stream) {
    (void)n_in; (void)ws_size;
    const float* x     = (const float*)d_in[0];
    const int*   eidx  = (const int*)d_in[1];
    const int*   batch = (const int*)d_in[2];
    const float* W1   = (const float*)d_in[4];
    const float* as1  = (const float*)d_in[5];
    const float* ad1  = (const float*)d_in[6];
    const float* b1   = (const float*)d_in[7];
    const float* lnw1 = (const float*)d_in[8];
    const float* lnb1 = (const float*)d_in[9];
    const float* W2   = (const float*)d_in[10];
    const float* as2  = (const float*)d_in[11];
    const float* ad2  = (const float*)d_in[12];
    const float* b2   = (const float*)d_in[13];
    const float* lnw2 = (const float*)d_in[14];
    const float* lnb2 = (const float*)d_in[15];
    const float* W3   = (const float*)d_in[16];
    const float* as3  = (const float*)d_in[17];
    const float* ad3  = (const float*)d_in[18];
    const float* b3   = (const float*)d_in[19];

    const int N    = in_sizes[0] / 128;
    const int E    = in_sizes[1] / 2;
    const int Etot = E + N;
    const int G    = out_size;
    const int* src  = eidx;
    const int* dstI = eidx + E;

    // workspace carve-up (256B aligned)
    char* w = (char*)d_ws;
    auto carve = [&](size_t bytes) {
        char* p = w;
        w += (bytes + 255) & ~(size_t)255;
        return p;
    };
    float*          bufA = (float*)carve((size_t)N * 128 * 4);
    float*          bufB = (float*)carve((size_t)N * 128 * 4);
    float*          asb  = (float*)carve((size_t)N * 4 * 4);
    float*          adb  = (float*)carve((size_t)N * 4 * 4);
    unsigned*       mb   = (unsigned*)carve((size_t)N * 4 * 4);
    float*          sb   = (float*)carve((size_t)N * 4 * 4);
    float*          ebuf = (float*)carve((size_t)Etot * 4 * 4);
    unsigned short* wt   = (unsigned short*)carve((size_t)128 * 128 * 2);
    float*          xp3  = (float*)carve((size_t)N * 4);
    float*          out3 = (float*)carve((size_t)N * 4);
    float*          pool = (float*)carve((size_t)2 * G * 4);

    const int edgeBlocks = (Etot + 255) / 256;
    const int nodeWaveBlocks = (N + 7) / 8;          // 8 waves per 256-thread block
    const int aggrBlocks = (Etot + 7) / 8;
    const int gemmBlocks = (N + 31) / 32;

    auto zero = [&](void* p, size_t bytes) {
        long long n4 = (long long)(bytes / 16);
        int blocks = (int)((n4 + 255) / 256);
        if (blocks > 4096) blocks = 4096;
        if (blocks < 1) blocks = 1;
        zero4_kernel<<<blocks, 256, 0, stream>>>((uint4*)p, n4);
    };

    // ---------------- layer 1: x -> bufA (GEMM) -> bufB (aggr) ----------------
    wt_convert_kernel<<<64, 256, 0, stream>>>(W1, wt);
    gemm_wmma_kernel<<<gemmBlocks, 256, 0, stream>>>(x, wt, bufA, N);
    zero(bufB, (size_t)N * 128 * 4);
    zero(mb, (size_t)N * 16);
    zero(sb, (size_t)N * 16);
    attn_coef_kernel<<<nodeWaveBlocks, 256, 0, stream>>>(bufA, as1, ad1, asb, adb, N);
    edge_max_kernel<<<edgeBlocks, 256, 0, stream>>>(src, dstI, asb, adb, mb, E, Etot);
    edge_exp_kernel<<<edgeBlocks, 256, 0, stream>>>(src, dstI, asb, adb, mb, ebuf, sb, E, Etot);
    edge_aggr_kernel<<<aggrBlocks, 256, 0, stream>>>(src, dstI, bufA, ebuf, sb, bufB, E, Etot);
    ln_relu_kernel<<<nodeWaveBlocks, 256, 0, stream>>>(bufB, b1, lnw1, lnb1, N);

    // ---------------- layer 2: bufB -> bufA (GEMM) -> bufB (aggr) -------------
    wt_convert_kernel<<<64, 256, 0, stream>>>(W2, wt);
    gemm_wmma_kernel<<<gemmBlocks, 256, 0, stream>>>(bufB, wt, bufA, N);
    zero(bufB, (size_t)N * 128 * 4);
    zero(mb, (size_t)N * 16);
    zero(sb, (size_t)N * 16);
    attn_coef_kernel<<<nodeWaveBlocks, 256, 0, stream>>>(bufA, as2, ad2, asb, adb, N);
    edge_max_kernel<<<edgeBlocks, 256, 0, stream>>>(src, dstI, asb, adb, mb, E, Etot);
    edge_exp_kernel<<<edgeBlocks, 256, 0, stream>>>(src, dstI, asb, adb, mb, ebuf, sb, E, Etot);
    edge_aggr_kernel<<<aggrBlocks, 256, 0, stream>>>(src, dstI, bufA, ebuf, sb, bufB, E, Etot);
    ln_relu_kernel<<<nodeWaveBlocks, 256, 0, stream>>>(bufB, b2, lnw2, lnb2, N);

    // ---------------- layer 3 (H=1, OUT=1) + pooling --------------------------
    zero(mb, (size_t)N * 4);
    zero(sb, (size_t)N * 4);
    zero(out3, (size_t)N * 4);
    zero(pool, (size_t)2 * G * 4);
    xp3_kernel<<<nodeWaveBlocks, 256, 0, stream>>>(bufB, W3, as3, ad3, xp3, asb, adb, N);
    edge_max1_kernel<<<edgeBlocks, 256, 0, stream>>>(src, dstI, asb, adb, mb, E, Etot);
    edge_exp1_kernel<<<edgeBlocks, 256, 0, stream>>>(src, dstI, asb, adb, mb, ebuf, sb, E, Etot);
    edge_aggr1_kernel<<<edgeBlocks, 256, 0, stream>>>(src, dstI, xp3, ebuf, sb, out3, E, Etot);
    pool_accum_kernel<<<(N + 255) / 256, 256, 0, stream>>>(out3, b3, batch, pool, N, G);
    pool_final_kernel<<<1, 256, 0, stream>>>(pool, (float*)d_out, G);
}
